// Attention_89816356094768
// MI455X (gfx1250) — matrix operand order repaired and checked by reading, not compiled, not run on hardware
//
#include <hip/hip_runtime.h>
#include <hip/hip_bf16.h>

typedef _Float16 half_t;
typedef __attribute__((ext_vector_type(16))) _Float16 v16h;
typedef __attribute__((ext_vector_type(8)))  _Float16 v8h;
typedef __attribute__((ext_vector_type(8)))  float    v8f;
typedef __attribute__((ext_vector_type(4)))  int      v4i;

#define T_SEQ 4096
#define HIDN  2048
#define NQH   16
#define NKVH  2
#define HD    128
#define QKV_COLS 2560              // 2048 q + 256 k + 256 v
#define ATT_SCALE 0.08838834764831845f   // 128^-0.5 (folded into Q at RoPE time)

// ---------------------------------------------------------------------------
// CDNA5 async global->LDS path (ASYNCcnt), with sync fallback.
// Builtin signature (from hipcc diagnostic): (as1 int4*, as3 int4*, imm, imm).
// ---------------------------------------------------------------------------
#if defined(__HIP_DEVICE_COMPILE__) && __has_builtin(__builtin_amdgcn_global_load_async_to_lds_b128)
#define HAVE_ASYNC_LDS 1
#else
#define HAVE_ASYNC_LDS 0
#endif

static __device__ __forceinline__ void async_copy16(const half_t* g, half_t* l) {
#if HAVE_ASYNC_LDS
  __builtin_amdgcn_global_load_async_to_lds_b128(
      (__attribute__((address_space(1))) v4i*)g,
      (__attribute__((address_space(3))) v4i*)l, 0, 0);
#else
  *(v8h*)l = *(const v8h*)g;
#endif
}

template <int N>
static __device__ __forceinline__ void wait_async() {
#if HAVE_ASYNC_LDS
#if __has_builtin(__builtin_amdgcn_s_wait_asynccnt)
  __builtin_amdgcn_s_wait_asynccnt((unsigned short)N);
#else
  asm volatile("s_wait_asynccnt %0" :: "n"(N) : "memory");
#endif
#endif
}

// ---------------------------------------------------------------------------
// WMMA fragment loaders per CDNA5 ISA 7.12.2 (wave32, 16-bit operands).
// A (16x32, MxK): lane L -> m=L&15, h=L>>4; halves e<8: K=h*8+e ; e>=8: K=16+h*8+(e-8)
// B (32x16, KxN): lane L -> n=L&15, g=L>>4; halves e<8: K=g*8+e ; e>=8: K=16+g*8+(e-8)
// C/D (16x16 f32): VGPR r, lane L -> m = r + 8*(L>>4), n = L&15.
// ---------------------------------------------------------------------------
static __device__ __forceinline__ v16h frag_a16(const half_t* base, int ld) {
  const int lane = threadIdx.x & 31;
  const int m = lane & 15, h = lane >> 4;
  const half_t* p = base + m * ld + h * 8;
  union { v16h v; v8h q[2]; } u;
  u.q[0] = *(const v8h*)p;
  u.q[1] = *(const v8h*)(p + 16);
  return u.v;
}

static __device__ __forceinline__ v16h frag_b16(const half_t* base, int ld) {
  const int lane = threadIdx.x & 31;
  const int n = lane & 15, g = lane >> 4;
  const half_t* p = base + n * ld + g * 8;
  union { v16h v; v8h q[2]; } u;
  u.q[0] = *(const v8h*)p;
  u.q[1] = *(const v8h*)(p + 16);
  return u.v;
}

// ---------------------------------------------------------------------------
// One-time f32 -> f16 conversion (8 elements / thread).
// ---------------------------------------------------------------------------
__global__ void cvt_f32_f16(const float* __restrict__ s, half_t* __restrict__ d, int n8) {
  const int i = blockIdx.x * blockDim.x + threadIdx.x;
  if (i >= n8) return;
  const float4 a = ((const float4*)s)[2 * i];
  const float4 b = ((const float4*)s)[2 * i + 1];
  v8h o;
  o[0] = (half_t)a.x; o[1] = (half_t)a.y; o[2] = (half_t)a.z; o[3] = (half_t)a.w;
  o[4] = (half_t)b.x; o[5] = (half_t)b.y; o[6] = (half_t)b.z; o[7] = (half_t)b.w;
  *(v8h*)(d + (size_t)i * 8) = o;
}

// ---------------------------------------------------------------------------
// C[M,N] = A[M,K] @ B[N,K]^T (+ bias), all-f16 operands, OutT in {f16, f32}.
// 128x128x32 tile, 8 waves (4x2), 8 WMMAs/wave/k-step, double-buffered LDS
// staged with GLOBAL_LOAD_ASYNC_TO_LDS_B128.
// ---------------------------------------------------------------------------
#define BM 128
#define BN 128
#define BK 32
#define LDK 40   // padded LDS stride in halves (multiple of 8 -> 16B aligned)

template <typename OutT>
__global__ __launch_bounds__(256) void gemm_bias_wmma(
    const half_t* __restrict__ A, const half_t* __restrict__ B,
    const float* __restrict__ bias, OutT* __restrict__ C,
    int M, int N, int K, int ldc) {
  __shared__ __align__(16) half_t As[2][BM * LDK];
  __shared__ __align__(16) half_t Bs[2][BN * LDK];
  const int tid  = threadIdx.x;
  const int wave = tid >> 5;
  const int lane = tid & 31;
  const int wm = wave >> 1;          // 0..3 : 32-row strip
  const int wn = wave & 1;           // 0..1 : 64-col strip
  const int row0 = blockIdx.y * BM;
  const int col0 = blockIdx.x * BN;

  const v8f vzero = {0.f,0.f,0.f,0.f,0.f,0.f,0.f,0.f};
  v8f acc[2][4];
#pragma unroll
  for (int i = 0; i < 2; ++i)
#pragma unroll
    for (int j = 0; j < 4; ++j) acc[i][j] = vzero;

  // stage one 128x32 f16 tile pair: 512 x 16B chunks each => 2 async ops per
  // thread per matrix => 4 async instructions per wave per tile.
  auto stage = [&](int k0, half_t* Ad, half_t* Bd) {
#pragma unroll
    for (int j = 0; j < 2; ++j) {
      const int f = tid + j * 256;   // 0..511
      const int r = f >> 2, c = f & 3;
      async_copy16(A + (size_t)(row0 + r) * K + k0 + c * 8, Ad + r * LDK + c * 8);
      async_copy16(B + (size_t)(col0 + r) * K + k0 + c * 8, Bd + r * LDK + c * 8);
    }
  };

  const int nk = K / BK;
  stage(0, As[0], Bs[0]);
  for (int kt = 0; kt < nk; ++kt) {
    const int cur = kt & 1;
    if (kt + 1 < nk) {
      stage((kt + 1) * BK, As[cur ^ 1], Bs[cur ^ 1]);
      wait_async<4>();   // older 4 (current tile) complete in-order
    } else {
      wait_async<0>();
    }
    __syncthreads();     // all waves' current tile visible

    const v16h a0 = frag_a16(As[cur] + (wm * 32)      * LDK, LDK);
    const v16h a1 = frag_a16(As[cur] + (wm * 32 + 16) * LDK, LDK);
#pragma unroll
    for (int jn = 0; jn < 4; ++jn) {
      const v16h b = frag_b16(Bs[cur] + (wn * 64 + jn * 16) * LDK, LDK);
      acc[0][jn] = __builtin_amdgcn_wmma_f32_16x16x32_f16(false, a0, false, b, (short)0, acc[0][jn], false, false);
      acc[1][jn] = __builtin_amdgcn_wmma_f32_16x16x32_f16(false, a1, false, b, (short)0, acc[1][jn], false, false);
    }
    __syncthreads();     // done reading buf[cur] before it is refilled
  }

  const int mh = lane >> 4, nn = lane & 15;
#pragma unroll
  for (int im = 0; im < 2; ++im)
#pragma unroll
    for (int jn = 0; jn < 4; ++jn) {
      const int col = col0 + wn * 64 + jn * 16 + nn;
      const float bb = bias ? bias[col] : 0.f;
#pragma unroll
      for (int r = 0; r < 8; ++r) {
        const int row = row0 + wm * 32 + im * 16 + r + 8 * mh;
        C[(size_t)row * ldc + col] = (OutT)(acc[im][jn][r] + bb);
      }
    }
}

// ---------------------------------------------------------------------------
// RoPE on q/k (f16 in, f16 out), relayout to [head][t][d]; folds the 1/sqrt(D)
// attention scale into Q.
// ---------------------------------------------------------------------------
__global__ void rope_convert(const half_t* __restrict__ qkv,
                             const float* __restrict__ cosT,
                             const float* __restrict__ sinT,
                             half_t* __restrict__ q16,
                             half_t* __restrict__ k16,
                             half_t* __restrict__ v16) {
  const int idx = blockIdx.x * blockDim.x + threadIdx.x;
  if (idx >= T_SEQ * QKV_COLS) return;
  const int t = idx / QKV_COLS;
  const int c = idx - t * QKV_COLS;
  const float val = (float)qkv[idx];
  if (c < NQH * HD) {
    const int hd = c >> 7, d = c & 127;
    const float cs = cosT[t * HD + d], sn = sinT[t * HD + d];
    const int od = (d < 64) ? d + 64 : d - 64;
    const float other = (float)qkv[(size_t)t * QKV_COLS + hd * HD + od];
    const float r = (d < 64) ? (val * cs - other * sn) : (val * cs + other * sn);
    q16[((size_t)hd * T_SEQ + t) * HD + d] = (half_t)(r * ATT_SCALE);
  } else if (c < NQH * HD + NKVH * HD) {
    const int c2 = c - NQH * HD;
    const int hd = c2 >> 7, d = c2 & 127;
    const float cs = cosT[t * HD + d], sn = sinT[t * HD + d];
    const int od = (d < 64) ? d + 64 : d - 64;
    const float other = (float)qkv[(size_t)t * QKV_COLS + NQH * HD + hd * HD + od];
    const float r = (d < 64) ? (val * cs - other * sn) : (val * cs + other * sn);
    k16[((size_t)hd * T_SEQ + t) * HD + d] = (half_t)r;
  } else {
    const int c2 = c - NQH * HD - NKVH * HD;
    const int hd = c2 >> 7, d = c2 & 127;
    v16[((size_t)hd * T_SEQ + t) * HD + d] = (half_t)val;
  }
}

// ---------------------------------------------------------------------------
// Flash attention (causal, GQA). Block = (q head, 128-query tile), 8 waves;
// each wave owns 16 query rows, loops 32-key blocks to the causal diagonal.
// QK^T: 8 WMMAs/step; PV: 8 WMMAs/step against V transposed in LDS.
// K tile staged via async global->LDS; V transposed manually (async can't).
// ---------------------------------------------------------------------------
#define LDW 136   // K tile LDS stride [key][d]
#define LDV 40    // Vt / P LDS stride

__global__ __launch_bounds__(256) void flash_attn_wmma(
    const half_t* __restrict__ q16, const half_t* __restrict__ k16,
    const half_t* __restrict__ v16, half_t* __restrict__ attn_out) {
  __shared__ __align__(16) half_t Ks[32 * LDW];       // [key][d]
  __shared__ __align__(16) half_t Vs[HD * LDV];       // [d][key]  (transposed)
  __shared__ __align__(16) half_t Ps[8][16 * LDV];    // per-wave P tile [m][key]

  const int qh  = blockIdx.x;       // 0..15
  const int qt  = blockIdx.y;       // 0..31
  const int kvh = qh >> 3;          // GROUPS = 8
  const int tid = threadIdx.x, wave = tid >> 5, lane = tid & 31;
  const int mh = lane >> 4, nn = lane & 15;
  const int rows0 = qt * 128 + wave * 16;

  const half_t* Qbase = q16 + ((size_t)qh * T_SEQ + rows0) * HD;
  v16h qa[4];
#pragma unroll
  for (int c = 0; c < 4; ++c) qa[c] = frag_a16(Qbase + c * 32, HD);

  const v8f vzero = {0.f,0.f,0.f,0.f,0.f,0.f,0.f,0.f};
  v8f o[8];
#pragma unroll
  for (int i = 0; i < 8; ++i) o[i] = vzero;
  float row_m[8], row_l[8];
#pragma unroll
  for (int r = 0; r < 8; ++r) { row_m[r] = -__builtin_inff(); row_l[r] = 0.f; }

  const half_t* Kg0 = k16 + (size_t)kvh * T_SEQ * HD;
  const half_t* Vg0 = v16 + (size_t)kvh * T_SEQ * HD;
  const int nkb = qt * 4 + 4;   // key blocks covering causal range of this tile

  for (int kb = 0; kb < nkb; ++kb) {
    const int s0 = kb * 32;
    __syncthreads();
#pragma unroll
    for (int j = 0; j < 2; ++j) {
      const int cc = tid + j * 256;     // 0..511
      const int ki = cc >> 4, dc = cc & 15;
      async_copy16(Kg0 + (size_t)(s0 + ki) * HD + dc * 8, Ks + ki * LDW + dc * 8);
      const v8h vv = *(const v8h*)(Vg0 + (size_t)(s0 + ki) * HD + dc * 8);
#pragma unroll
      for (int e = 0; e < 8; ++e) Vs[(dc * 8 + e) * LDV + ki] = vv[e];
    }
    wait_async<0>();   // this wave's K async copies landed in LDS
    __syncthreads();

    // S = Q @ K^T for two 16-key tiles (scale pre-folded into Q)
    v8f s[2]; s[0] = vzero; s[1] = vzero;
#pragma unroll
    for (int c = 0; c < 4; ++c) {
#pragma unroll
      for (int j = 0; j < 2; ++j) {
        const v16h kf = frag_b16(Ks + (j * 16) * LDW + c * 32, LDW);
        s[j] = __builtin_amdgcn_wmma_f32_16x16x32_f16(false, qa[c], false, kf, (short)0, s[j], false, false);
      }
    }

    // causal mask + online softmax (row reductions within 16-lane halves)
#pragma unroll
    for (int r = 0; r < 8; ++r) {
      const int trow = rows0 + r + 8 * mh;
      float v0 = s[0][r];
      float v1 = s[1][r];
      if (s0 + nn > trow)      v0 = -__builtin_inff();
      if (s0 + 16 + nn > trow) v1 = -__builtin_inff();
      float mx = fmaxf(v0, v1);
#pragma unroll
      for (int w = 8; w >= 1; w >>= 1) mx = fmaxf(mx, __shfl_xor(mx, w, 16));
      const float mnew = fmaxf(row_m[r], mx);
      const float corr = __expf(row_m[r] - mnew);
      row_m[r] = mnew;
      const float p0 = __expf(v0 - mnew);
      const float p1 = __expf(v1 - mnew);
      float ps = p0 + p1;
#pragma unroll
      for (int w = 8; w >= 1; w >>= 1) ps += __shfl_xor(ps, w, 16);
      row_l[r] = row_l[r] * corr + ps;
#pragma unroll
      for (int t8 = 0; t8 < 8; ++t8) o[t8][r] *= corr;
      const int pm = r + 8 * mh;
      Ps[wave][pm * LDV + nn]      = (half_t)p0;
      Ps[wave][pm * LDV + 16 + nn] = (half_t)p1;
    }

    // O += P(16x32) @ V(32x128); per-wave LDS is in-order within the wave
    const v16h pa = frag_a16(&Ps[wave][0], LDV);
#pragma unroll
    for (int dc = 0; dc < 8; ++dc) {
      const v16h vb = frag_b16(Vs + (dc * 16) * LDV, LDV);
      o[dc] = __builtin_amdgcn_wmma_f32_16x16x32_f16(false, pa, false, vb, (short)0, o[dc], false, false);
    }
  }

  // normalize + store f16 [t][qh*128 + d] for the wo GEMM
#pragma unroll
  for (int r = 0; r < 8; ++r) {
    const int trow = rows0 + r + 8 * mh;
    const float inv = 1.f / row_l[r];
#pragma unroll
    for (int dc = 0; dc < 8; ++dc) {
      attn_out[(size_t)trow * (NQH * HD) + qh * HD + dc * 16 + nn] =
          (half_t)(o[dc][r] * inv);
    }
  }
}

// ---------------------------------------------------------------------------
extern "C" void kernel_launch(void* const* d_in, const int* in_sizes, int n_in,
                              void* d_out, int out_size, void* d_ws, size_t ws_size,
                              hipStream_t stream) {
  (void)in_sizes; (void)n_in; (void)out_size; (void)ws_size;
  const float* x    = (const float*)d_in[0];
  const float* cosT = (const float*)d_in[1];
  const float* sinT = (const float*)d_in[2];
  const float* wq   = (const float*)d_in[3];
  const float* bq   = (const float*)d_in[4];
  const float* wk   = (const float*)d_in[5];
  const float* bk   = (const float*)d_in[6];
  const float* wv   = (const float*)d_in[7];
  const float* bv   = (const float*)d_in[8];
  const float* wo   = (const float*)d_in[9];
  float* out = (float*)d_out;

  char* ws = (char*)d_ws;
  size_t off = 0;
  half_t* qkv16 = (half_t*)(ws + off); off += (size_t)T_SEQ * QKV_COLS * 2;  // reused as attn16
  half_t* x16   = (half_t*)(ws + off); off += (size_t)T_SEQ * HIDN * 2;
  half_t* wq16  = (half_t*)(ws + off); off += (size_t)HIDN * HIDN * 2;
  half_t* wk16  = (half_t*)(ws + off); off += (size_t)(NKVH * HD) * HIDN * 2;
  half_t* wv16  = (half_t*)(ws + off); off += (size_t)(NKVH * HD) * HIDN * 2;
  half_t* wo16  = (half_t*)(ws + off); off += (size_t)HIDN * HIDN * 2;
  half_t* q16   = (half_t*)(ws + off); off += (size_t)NQH  * T_SEQ * HD * 2;
  half_t* k16   = (half_t*)(ws + off); off += (size_t)NKVH * T_SEQ * HD * 2;
  half_t* v16   = (half_t*)(ws + off);
  half_t* attn16 = qkv16;  // safe reuse: rope consumes qkv16 before attention writes

  const dim3 blk(256);
  // one-time f16 conversion of activations and weights
  cvt_f32_f16<<<(T_SEQ * HIDN / 8 + 255) / 256, blk, 0, stream>>>(x,  x16,  T_SEQ * HIDN / 8);
  cvt_f32_f16<<<(HIDN * HIDN / 8 + 255) / 256, blk, 0, stream>>>(wq, wq16, HIDN * HIDN / 8);
  cvt_f32_f16<<<(NKVH * HD * HIDN / 8 + 255) / 256, blk, 0, stream>>>(wk, wk16, NKVH * HD * HIDN / 8);
  cvt_f32_f16<<<(NKVH * HD * HIDN / 8 + 255) / 256, blk, 0, stream>>>(wv, wv16, NKVH * HD * HIDN / 8);
  cvt_f32_f16<<<(HIDN * HIDN / 8 + 255) / 256, blk, 0, stream>>>(wo, wo16, HIDN * HIDN / 8);

  // QKV projections (f16 in / f16 out, fused bias)
  gemm_bias_wmma<half_t><<<dim3(2048 / BN, 4096 / BM), blk, 0, stream>>>(x16, wq16, bq, qkv16,        4096, 2048, 2048, QKV_COLS);
  gemm_bias_wmma<half_t><<<dim3( 256 / BN, 4096 / BM), blk, 0, stream>>>(x16, wk16, bk, qkv16 + 2048, 4096,  256, 2048, QKV_COLS);
  gemm_bias_wmma<half_t><<<dim3( 256 / BN, 4096 / BM), blk, 0, stream>>>(x16, wv16, bv, qkv16 + 2304, 4096,  256, 2048, QKV_COLS);

  const int tot = T_SEQ * QKV_COLS;
  rope_convert<<<(tot + 255) / 256, blk, 0, stream>>>(qkv16, cosT, sinT, q16, k16, v16);

  flash_attn_wmma<<<dim3(NQH, T_SEQ / 128), blk, 0, stream>>>(q16, k16, v16, attn16);

  // output projection -> f32 d_out
  gemm_bias_wmma<float><<<dim3(2048 / BN, 4096 / BM), blk, 0, stream>>>(attn16, wo16, nullptr, out, 4096, 2048, 2048, 2048);
}
